// STGCNBlock_82008105550537
// MI455X (gfx1250) — matrix operand order repaired and checked by reading, not compiled, not run on hardware
//
#include <hip/hip_runtime.h>
#include <hip/hip_bf16.h>
#include <stdint.h>

// ---------------- problem constants ----------------
#define BATCH   64
#define CIN     64
#define COUT    128
#define T_DIM   300
#define V_DIM   25
#define NPIX_B  (T_DIM * V_DIM)          // 7500 pixels per batch sample
#define NP      (BATCH * NPIX_B)         // 480000 total pixels
#define NPT     (NP / 16)                // 30000 16-pixel tiles
#define KT      3                        // temporal kernel size
#define BN_EPS  1e-5f

typedef __attribute__((ext_vector_type(16))) _Float16     v16h;
typedef __attribute__((ext_vector_type(8)))  _Float16     v8h;
typedef __attribute__((ext_vector_type(8)))  float        v8f;
typedef __attribute__((ext_vector_type(4)))  unsigned int v4u;
typedef __attribute__((ext_vector_type(8)))  int          v8i;
typedef __attribute__((ext_vector_type(4)))  int          v4i;

#if defined(__AMDGCN__) && __has_builtin(__builtin_amdgcn_tensor_load_to_lds)
#define USE_TDM 1
#else
#define USE_TDM 0
#endif

// ----------------------------------------------------------------------------
// WMMA fragment maps (CDNA5 ISA 7.12.2, wave32):
//  A (16x32 f16): lane L: M = L&15 ; element e -> K = e + ((L&16)?16:0) (contiguous order chosen here)
//  B (32x16 f16): lane L: N = L&15 ; element e -> K = e + ((L&16)?16:0) (same order as A, chosen here)
//  C/D (16x16 f32): lane L: N = L&15 ; vgpr r -> M = r + ((L&16)?8:0)
// B-operand buffers (xh, support) stored pre-swizzled in fragment order:
//  offset(c,P) = ((cb*NPT + P/16)*32 + lane)*16 + e  with cb=c>>5, kk=c&31,
//  lane = (P&15) + (kk&16), e = kk&15  -> each lane's 16 halves contiguous.
// A-operand weights pre-packed identically by pack_weights.
// ----------------------------------------------------------------------------
__device__ __forceinline__ size_t swz_off(int c, int P) {
    int cb = c >> 5, kk = c & 31;
    int pt = P >> 4, n = P & 15;
    int lane = n + (kk & 16);
    int e = kk & 15;
    return ((((size_t)cb * NPT) + pt) * 32 + lane) * 16 + e;
}

__device__ __forceinline__ int a_kmap(int e, int hi) {
    return e + (hi ? 16 : 0);
}

// ----------------------------------------------------------------------------
// TDM helper: load a (512 x nrow) f16 tile (rows strided by `stride` halves)
// from global into LDS at lds_byte_off.  Issued by the calling wave (one
// tensor op per workgroup); caller is wave 0 and then waits TENSORcnt==0.
// ----------------------------------------------------------------------------
#if USE_TDM
__device__ __forceinline__ void tdm_load_rows_512(const _Float16* gsrc,
                                                  unsigned lds_byte_off,
                                                  int nrow, long long stride_halves) {
    unsigned long long ga = (unsigned long long)(uintptr_t)gsrc;
    v4u g0;
    g0.x = 1u;                                   // count=1, user descriptor
    g0.y = lds_byte_off;                         // lds_addr
    g0.z = (unsigned)ga;                         // global_addr[31:0]
    g0.w = (unsigned)((ga >> 32) & 0x01FFFFFFu) | 0x80000000u; // addr[56:32] | type=2
    v8i g1;
    g1[0] = 0x10000;                             // workgroup_mask=0, data_size=1 (2B)
    g1[1] = (int)(512u << 16);                   // tensor_dim0 = 512
    g1[2] = (int)((unsigned)nrow << 16);         // tensor_dim1 = nrow
    g1[3] = (int)(512u << 16);                   // tile_dim0 = 512
    g1[4] = nrow;                                // tile_dim1 = nrow, tile_dim2 = 0
    g1[5] = (int)(stride_halves & 0xFFFFFFFF);   // tensor_dim0_stride[31:0]
    g1[6] = (int)((stride_halves >> 32) & 0xFFFF);
    g1[7] = 0;
    v4i gz = {0, 0, 0, 0};
#if __clang_major__ >= 23
    v8i gz8 = {0, 0, 0, 0, 0, 0, 0, 0};
    __builtin_amdgcn_tensor_load_to_lds(g0, g1, gz, gz, gz8, 0);
#else
    __builtin_amdgcn_tensor_load_to_lds(g0, g1, gz, gz, 0);
#endif
}
#endif

// ============================================================================
// Pass 0: pack all weights to f16 in A-fragment lane order (one-shot, 64K
// elements).  wsp/wrp: [otile][cb(2)][512]; wtp: [otile][s*4+cb(12)][512].
// ============================================================================
__global__ void pack_weights(const float* __restrict__ Ws,
                             const float* __restrict__ Wr,
                             const float* __restrict__ Wt,
                             _Float16* __restrict__ wsp,
                             _Float16* __restrict__ wrp,
                             _Float16* __restrict__ wtp) {
    int i = blockIdx.x * blockDim.x + threadIdx.x;
    int local = i & 511;
    int e = local & 15, lane = local >> 4;
    int m = lane & 15, hi = (lane >> 4) & 1;
    int kk = a_kmap(e, hi);
    if (i < 8192) {                               // W_s fragments
        int f = i >> 9, ot = f >> 1, cb = f & 1;
        wsp[i] = (_Float16)Ws[(ot * 16 + m) * CIN + cb * 32 + kk];
    } else if (i < 16384) {                       // W_r fragments
        int j = i - 8192;
        int f = j >> 9, ot = f >> 1, cb = f & 1;
        wrp[j] = (_Float16)Wr[(ot * 16 + m) * CIN + cb * 32 + kk];
    } else if (i < 65536) {                       // W_t fragments
        int j = i - 16384;
        int f = j >> 9, ot = f / 12, sc = f % 12;
        int s = sc >> 2, cb = sc & 3;
        wtp[j] = (_Float16)Wt[((ot * 16 + m) * COUT + cb * 32 + kk) * KT + s];
    }
}

// ============================================================================
// Pass 1: support = x @ A over joints (K=25); pack x + support to f16 in
// B-fragment swizzled layout.  One thread per (b,c,t) row of 25 joints.
// ============================================================================
__global__ void graph_conv_pack(const float* __restrict__ x,
                                const float* __restrict__ Ag,
                                _Float16* __restrict__ xh,
                                _Float16* __restrict__ suph) {
    __shared__ float As[V_DIM * V_DIM];
    for (int i = threadIdx.x; i < V_DIM * V_DIM; i += blockDim.x) As[i] = Ag[i];
    __syncthreads();

    int row = blockIdx.x * blockDim.x + threadIdx.x;     // (b*CIN + c)*T + t
    if (row >= BATCH * CIN * T_DIM) return;
    int t  = row % T_DIM;
    int bc = row / T_DIM;
    int c  = bc % CIN;
    int b  = bc / CIN;

    const float* xr = x + (size_t)row * V_DIM;
    float xv[V_DIM];
#pragma unroll
    for (int n = 0; n < V_DIM; ++n) xv[n] = xr[n];

    int Pbase = b * NPIX_B + t * V_DIM;
#pragma unroll
    for (int v = 0; v < V_DIM; ++v) {
        float s = 0.f;
#pragma unroll
        for (int n = 0; n < V_DIM; ++n) s = fmaf(xv[n], As[n * V_DIM + v], s);
        int P = Pbase + v;
        size_t off = swz_off(c, P);
        xh[off]   = (_Float16)xv[v];
        suph[off] = (_Float16)s;
    }
}

// ============================================================================
// Pass 2: ys = W_s(128x64) @ support(64xNP).  One block per 16-pixel tile,
// 8 waves cover all 128 output channels.  B operand staged by TDM.
// Accumulates BN1 sum/sumsq.  (b_s omitted: cancels in training-mode BN.)
// ============================================================================
__global__ void spatial_gemm(const _Float16* __restrict__ suph,
                             const _Float16* __restrict__ wsp,
                             _Float16* __restrict__ ysh,
                             float* __restrict__ stat_sum,
                             float* __restrict__ stat_sq) {
    __shared__ __align__(32) _Float16 Bsm[2 * 512];   // 2 K-blocks of B frags
    int ntile = blockIdx.x;
    int p0 = ntile * 16;

#if USE_TDM
    if (threadIdx.x < 32) {      // wave 0 issues one TDM op: 512x2 tile
        tdm_load_rows_512(suph + (size_t)ntile * 512,
                          (unsigned)(uintptr_t)&Bsm[0],
                          2, (long long)NPT * 512);
        __builtin_amdgcn_s_wait_tensorcnt(0);
    }
#else
    {
        int cb = threadIdx.x >> 7;
        int q  = threadIdx.x & 127;
        *(uint64_t*)&Bsm[cb * 512 + q * 4] =
            *(const uint64_t*)(suph + ((size_t)cb * NPT + ntile) * 512 + q * 4);
    }
#endif
    __syncthreads();

    int wave = threadIdx.x >> 5, lane = threadIdx.x & 31;
    int o0 = wave * 16;

    v8f acc = {};
#pragma unroll
    for (int cb = 0; cb < 2; ++cb) {
        v16h af = *(const v16h*)(wsp + ((size_t)(wave * 2 + cb)) * 512 + lane * 16);
        v16h bf = *(const v16h*)&Bsm[cb * 512 + lane * 16];
        acc = __builtin_amdgcn_wmma_f32_16x16x32_f16(false, af, false, bf,
                                                     (short)0, acc, false, false);
    }

    int n = lane & 15, hi = (lane >> 4) & 1;
#pragma unroll
    for (int r = 0; r < 8; ++r) {
        int o = o0 + r + 8 * hi;
        float v = acc[r];
        ysh[(size_t)o * NP + p0 + n] = (_Float16)v;
        float s1 = v, s2 = v * v;
#pragma unroll
        for (int off = 1; off < 16; off <<= 1) {
            s1 += __shfl_xor(s1, off, 32);
            s2 += __shfl_xor(s2, off, 32);
        }
        if (n == 0) {
            atomicAdd(&stat_sum[o], s1);
            atomicAdd(&stat_sq[o],  s2);
        }
    }
}

// ============================================================================
// BN finalize: scale/shift per channel from accumulated sum / sumsq.
// ============================================================================
__global__ void bn_finalize(const float* __restrict__ ssum,
                            const float* __restrict__ ssq,
                            const float* __restrict__ gamma,
                            const float* __restrict__ beta,
                            float* __restrict__ scale,
                            float* __restrict__ shift) {
    int c = threadIdx.x;
    if (c < COUT) {
        float mean = ssum[c] * (1.0f / NP);
        float var  = ssq[c]  * (1.0f / NP) - mean * mean;
        float sc = gamma[c] * rsqrtf(var + BN_EPS);
        scale[c] = sc;
        shift[c] = beta[c] - mean * sc;
    }
}

// ============================================================================
// BN1 + ReLU applied in place to f16 activations (8 halves / thread).
// ============================================================================
__global__ void bn_relu_inplace(_Float16* __restrict__ ysh,
                                const float* __restrict__ scale,
                                const float* __restrict__ shift) {
    size_t i = ((size_t)blockIdx.x * blockDim.x + threadIdx.x) * 8;
    int o = (int)(i / NP);
    float sc = scale[o], sf = shift[o];
    v8h d = *(v8h*)(ysh + i);
#pragma unroll
    for (int j = 0; j < 8; ++j) {
        float f = (float)d[j] * sc + sf;
        d[j] = (_Float16)(f > 0.f ? f : 0.f);
    }
    *(v8h*)(ysh + i) = d;
}

// ============================================================================
// Pass 3: temporal conv as GEMM, M=128, K=384 (=128 ch x 3 taps), per-column
// T-boundary masking on the +/-1 shifts.  12 WMMAs per wave, pre-packed f16
// weight fragments.  z -> d_out, BN2 stats accumulated.  (b_t cancels in BN.)
// ============================================================================
__global__ void temporal_gemm(const _Float16* __restrict__ ah,
                              const _Float16* __restrict__ wtp,
                              float* __restrict__ z,
                              float* __restrict__ stat_sum,
                              float* __restrict__ stat_sq) {
    __shared__ __align__(32) _Float16 Bsm[12 * 512];  // [shift(3)][cblock(4)] frags
    int ntile = blockIdx.x;
    int p0 = ntile * 16;

    // cooperative gather of 12 B fragments with temporal shift + masking
    for (int f = threadIdx.x; f < 12 * 512; f += blockDim.x) {
        int frag = f >> 9;           // s*4 + cb
        int idx  = f & 511;
        int lane = idx >> 4, e = idx & 15;
        int s  = frag >> 2;          // tap 0..2 -> time shift s-1
        int cb = frag & 3;
        int kk = e + ((lane & 16) ? 16 : 0);
        int c  = cb * 32 + kk;
        int P  = p0 + (lane & 15);
        int t  = (P % NPIX_B) / V_DIM;
        int ts = t + s - 1;
        _Float16 val = (_Float16)0.f;
        if (ts >= 0 && ts < T_DIM)
            val = ah[(size_t)c * NP + P + (s - 1) * V_DIM];
        Bsm[f] = val;
    }
    __syncthreads();

    int wave = threadIdx.x >> 5, lane = threadIdx.x & 31;
    int o0 = wave * 16;
    const _Float16* wbase = wtp + (size_t)wave * 12 * 512 + lane * 16;

    v8f acc = {};
#pragma unroll
    for (int s = 0; s < KT; ++s) {
#pragma unroll
        for (int cb = 0; cb < 4; ++cb) {
            v16h af = *(const v16h*)(wbase + (size_t)(s * 4 + cb) * 512);
            v16h bf = *(const v16h*)&Bsm[(s * 4 + cb) * 512 + lane * 16];
            acc = __builtin_amdgcn_wmma_f32_16x16x32_f16(false, af, false, bf,
                                                         (short)0, acc, false, false);
        }
    }

    int n = lane & 15, hi = (lane >> 4) & 1;
    int P = p0 + n;
    int b = P / NPIX_B;
    int q = P % NPIX_B;
    size_t base = (size_t)b * (COUT * NPIX_B) + q;   // d_out is [b][o][t][v]
#pragma unroll
    for (int r = 0; r < 8; ++r) {
        int o = o0 + r + 8 * hi;
        float v = acc[r];
        z[base + (size_t)o * NPIX_B] = v;
        float s1 = v, s2 = v * v;
#pragma unroll
        for (int off = 1; off < 16; off <<= 1) {
            s1 += __shfl_xor(s1, off, 32);
            s2 += __shfl_xor(s2, off, 32);
        }
        if (n == 0) {
            atomicAdd(&stat_sum[o], s1);
            atomicAdd(&stat_sq[o],  s2);
        }
    }
}

// ============================================================================
// Pass 4: residual GEMM (W_r @ x, K=64) fused with BN2+ReLU on z (in d_out)
// and + b_r.  B operand staged by TDM.  In-place RMW of d_out.
// ============================================================================
__global__ void residual_final(const _Float16* __restrict__ xh,
                               const _Float16* __restrict__ wrp,
                               const float* __restrict__ br,
                               const float* __restrict__ scale2,
                               const float* __restrict__ shift2,
                               float* __restrict__ out) {
    __shared__ __align__(32) _Float16 Bsm[2 * 512];
    int ntile = blockIdx.x;
    int p0 = ntile * 16;

#if USE_TDM
    if (threadIdx.x < 32) {
        tdm_load_rows_512(xh + (size_t)ntile * 512,
                          (unsigned)(uintptr_t)&Bsm[0],
                          2, (long long)NPT * 512);
        __builtin_amdgcn_s_wait_tensorcnt(0);
    }
#else
    {
        int cb = threadIdx.x >> 7;
        int q  = threadIdx.x & 127;
        *(uint64_t*)&Bsm[cb * 512 + q * 4] =
            *(const uint64_t*)(xh + ((size_t)cb * NPT + ntile) * 512 + q * 4);
    }
#endif
    __syncthreads();

    int wave = threadIdx.x >> 5, lane = threadIdx.x & 31;
    int o0 = wave * 16;

    v8f acc = {};
#pragma unroll
    for (int cb = 0; cb < 2; ++cb) {
        v16h af = *(const v16h*)(wrp + ((size_t)(wave * 2 + cb)) * 512 + lane * 16);
        v16h bf = *(const v16h*)&Bsm[cb * 512 + lane * 16];
        acc = __builtin_amdgcn_wmma_f32_16x16x32_f16(false, af, false, bf,
                                                     (short)0, acc, false, false);
    }

    int n = lane & 15, hi = (lane >> 4) & 1;
    int P = p0 + n;
    int b = P / NPIX_B;
    int q = P % NPIX_B;
    size_t base = (size_t)b * (COUT * NPIX_B) + q;
#pragma unroll
    for (int r = 0; r < 8; ++r) {
        int o = o0 + r + 8 * hi;
        size_t idx = base + (size_t)o * NPIX_B;
        float zv = out[idx];
        float a = zv * scale2[o] + shift2[o];
        a = a > 0.f ? a : 0.f;
        out[idx] = a + acc[r] + br[o];
    }
}

// ============================================================================
// Launch
// ============================================================================
extern "C" void kernel_launch(void* const* d_in, const int* in_sizes, int n_in,
                              void* d_out, int out_size, void* d_ws, size_t ws_size,
                              hipStream_t stream) {
    const float* x       = (const float*)d_in[0];
    const float* A       = (const float*)d_in[1];
    const float* W_s     = (const float*)d_in[2];
    // d_in[3] = b_s   (cancels in BN, unused)
    const float* gamma_s = (const float*)d_in[4];
    const float* beta_s  = (const float*)d_in[5];
    const float* W_t     = (const float*)d_in[6];
    // d_in[7] = b_t   (cancels in BN, unused)
    const float* gamma_t = (const float*)d_in[8];
    const float* beta_t  = (const float*)d_in[9];
    const float* W_r     = (const float*)d_in[10];
    const float* b_r     = (const float*)d_in[11];
    float* out = (float*)d_out;

    // workspace carve-up
    const size_t XH_BYTES  = (size_t)CIN  * NP * 2;   // 61,440,000
    const size_t SUP_BYTES = (size_t)CIN  * NP * 2;   // 61,440,000
    const size_t YS_BYTES  = (size_t)COUT * NP * 2;   // 122,880,000
    char* ws = (char*)d_ws;
    _Float16* xh   = (_Float16*)ws;
    _Float16* suph = (_Float16*)(ws + XH_BYTES);
    _Float16* ysh  = (_Float16*)(ws + XH_BYTES + SUP_BYTES);
    float* stats   = (float*)(ws + XH_BYTES + SUP_BYTES + YS_BYTES);
    float* sum1 = stats,          *sq1 = stats + COUT;
    float* sc1  = stats + 2*COUT, *sh1 = stats + 3*COUT;
    float* sum2 = stats + 4*COUT, *sq2 = stats + 5*COUT;
    float* sc2  = stats + 6*COUT, *sh2 = stats + 7*COUT;
    _Float16* wsp = (_Float16*)(ws + XH_BYTES + SUP_BYTES + YS_BYTES + 4096);
    _Float16* wrp = wsp + 8192;
    _Float16* wtp = wrp + 8192;   // 49152 halves

    hipMemsetAsync(stats, 0, 8 * COUT * sizeof(float), stream);

    pack_weights<<<256, 256, 0, stream>>>(W_s, W_r, W_t, wsp, wrp, wtp);
    graph_conv_pack<<<(BATCH * CIN * T_DIM + 255) / 256, 256, 0, stream>>>(x, A, xh, suph);
    spatial_gemm<<<NPT, 256, 0, stream>>>(suph, wsp, ysh, sum1, sq1);
    bn_finalize<<<1, 128, 0, stream>>>(sum1, sq1, gamma_s, beta_s, sc1, sh1);
    bn_relu_inplace<<<(COUT * (NP / 8)) / 256, 256, 0, stream>>>(ysh, sc1, sh1);
    temporal_gemm<<<NPT, 256, 0, stream>>>(ysh, wtp, out, sum2, sq2);
    bn_finalize<<<1, 128, 0, stream>>>(sum2, sq2, gamma_t, beta_t, sc2, sh2);
    residual_final<<<NPT, 256, 0, stream>>>(xh, wrp, b_r, sc2, sh2, out);
}
